// Attention_33689723470464
// MI455X (gfx1250) — matrix operand order repaired and checked by reading, not compiled, not run on hardware
//
#include <hip/hip_runtime.h>
#include <hip/hip_bf16.h>

// ---------------------------------------------------------------------------
// CDNA5 (gfx1250) fused attention: bf16 WMMA 16x16x32, flash-style softmax.
// ---------------------------------------------------------------------------

typedef __attribute__((ext_vector_type(16))) __bf16 v16bf;
typedef __attribute__((ext_vector_type(8)))  __bf16 v8bf;
typedef __attribute__((ext_vector_type(8)))  float  v8f;

union BF16x16 { v16bf v; v8bf h[2]; };

#define NH 20
#define HD 64
#define CC 1280
#define TT 2048
#define BBATCH 4
#define MROWS (BBATCH * TT)   // 8192

__device__ __forceinline__ v8f wmma_bf16(const BF16x16& a, const BF16x16& b, v8f c) {
  // D = A(16x32 bf16) * B(32x16 bf16) + C(16x16 f32)
  return __builtin_amdgcn_wmma_f32_16x16x32_bf16(false, a.v, false, b.v,
                                                 (short)0, c, false, false);
}

// ---------------------------------------------------------------------------
// fp32 -> bf16 cast (grid-stride)
// ---------------------------------------------------------------------------
__global__ void cast_f32_bf16(const float* __restrict__ src,
                              __bf16* __restrict__ dst, int n) {
  int i = blockIdx.x * blockDim.x + threadIdx.x;
  int stride = gridDim.x * blockDim.x;
  for (; i < n; i += stride) dst[i] = (__bf16)src[i];
}

// ---------------------------------------------------------------------------
// 64x64-per-wave bf16 GEMM:  out[m,n] = sum_k X[m,k] * W[n,k]
//   mode 0: store bf16 head-split   [B,H,T,D]   (Q, K)
//   mode 1: store bf16 head-split-T [B,H,D,T]   (V)
//   mode 2: store f32 row-major [M,C] + bias    (final projection)
// ---------------------------------------------------------------------------
__global__ void __launch_bounds__(256)
gemm_bf16_64x64(const __bf16* __restrict__ X,   // [MROWS, CC]
                const __bf16* __restrict__ W,   // [CC, CC], row = out feature
                void* __restrict__ out,
                const float* __restrict__ bias,
                int mode) {
  const int lane = threadIdx.x & 31;
  const int warp = threadIdx.x >> 5;
  const int wid  = blockIdx.x * 8 + warp;     // 2560 waves total
  const int ntiles = CC / 64;                 // 20
  const int mtile  = wid / ntiles;            // 0..127
  const int ntile  = wid % ntiles;            // 0..19
  const int half = lane >> 4;
  const int l16  = lane & 15;
  const int m0 = mtile * 64;
  const int n0 = ntile * 64;

  v8f acc[4][4];
  v8f zero = {};
#pragma unroll
  for (int im = 0; im < 4; ++im)
#pragma unroll
    for (int jn = 0; jn < 4; ++jn) acc[im][jn] = zero;

  for (int k0 = 0; k0 < CC; k0 += 32) {
    BF16x16 a[4], b[4];
#pragma unroll
    for (int im = 0; im < 4; ++im) {
      const __bf16* p = X + (size_t)(m0 + im * 16 + l16) * CC + k0 + half * 8;
      a[im].h[0] = *(const v8bf*)(p);
      a[im].h[1] = *(const v8bf*)(p + 16);
      // unconditional: speculative prefetch of an OOB address is silently
      // dropped, and +64B stays inside the workspace here anyway.
      __builtin_prefetch(p + 32, 0, 1);       // global_prefetch_b8
    }
#pragma unroll
    for (int jn = 0; jn < 4; ++jn) {
      const __bf16* p = W + (size_t)(n0 + jn * 16 + l16) * CC + k0 + half * 8;
      b[jn].h[0] = *(const v8bf*)(p);
      b[jn].h[1] = *(const v8bf*)(p + 16);
      __builtin_prefetch(p + 32, 0, 1);
    }
#pragma unroll
    for (int im = 0; im < 4; ++im)
#pragma unroll
      for (int jn = 0; jn < 4; ++jn)
        acc[im][jn] = wmma_bf16(a[im], b[jn], acc[im][jn]);
  }

  // store: C/D layout => lane (n = l16, row-half = half), reg r => m = 8*half + r
#pragma unroll
  for (int im = 0; im < 4; ++im)
#pragma unroll
    for (int jn = 0; jn < 4; ++jn)
#pragma unroll
      for (int r = 0; r < 8; ++r) {
        const int m = m0 + im * 16 + half * 8 + r;
        const int n = n0 + jn * 16 + l16;
        const float v = acc[im][jn][r];
        if (mode == 2) {
          ((float*)out)[(size_t)m * CC + n] = v + bias[n];
        } else {
          const int bb = m >> 11, t = m & (TT - 1);  // T = 2048
          const int hh = n >> 6,  d = n & 63;        // D = 64
          size_t idx;
          if (mode == 0)
            idx = ((size_t)((bb * NH + hh) * TT + t)) * HD + d;
          else
            idx = ((size_t)((bb * NH + hh) * HD + d)) * TT + t;
          ((__bf16*)out)[idx] = (__bf16)v;
        }
      }
}

// ---------------------------------------------------------------------------
// Flash attention: one wave per (b, h, 64-query tile). D = 64, no mask.
//   Qh, Kh: [B,H,T,D] bf16    Vt: [B,H,D,T] bf16    Aout: [B,T,C] bf16
// ---------------------------------------------------------------------------
__global__ void __launch_bounds__(128)
flash_attn(const __bf16* __restrict__ Qh, const __bf16* __restrict__ Kh,
           const __bf16* __restrict__ Vt, __bf16* __restrict__ Aout) {
  __shared__ __bf16 Pb[4][64 * 72];   // per-wave P tile, padded stride (144B)

  const int lane = threadIdx.x & 31;
  const int warp = threadIdx.x >> 5;
  const int wid  = blockIdx.x * 4 + warp;   // 2560 waves
  const int qt = wid & 31;                  // 32 query tiles of 64
  const int bh = wid >> 5;                  // 0..79
  const int b = bh / NH, h = bh % NH;
  const int half = lane >> 4;
  const int l16  = lane & 15;
  const float scale = 0.125f;               // 1/sqrt(64)

  const __bf16* Qbase = Qh + (size_t)bh * TT * HD;
  const __bf16* Kbase = Kh + (size_t)bh * TT * HD;
  const __bf16* Vbase = Vt + (size_t)bh * HD * TT;
  __bf16* lds = Pb[warp];
  const int q0 = qt * 64;

  // Resident Q fragments: A-matrix 16x32, two K-steps cover D=64
  BF16x16 aQ[4][2];
#pragma unroll
  for (int im = 0; im < 4; ++im)
#pragma unroll
    for (int ks = 0; ks < 2; ++ks) {
      const __bf16* p = Qbase + (size_t)(q0 + im * 16 + l16) * HD + ks * 32 + half * 8;
      aQ[im][ks].h[0] = *(const v8bf*)(p);
      aQ[im][ks].h[1] = *(const v8bf*)(p + 16);
    }

  v8f O[4][4];
  v8f zero = {};
  float rowm[4][8], rowl[4][8];
#pragma unroll
  for (int im = 0; im < 4; ++im) {
#pragma unroll
    for (int jd = 0; jd < 4; ++jd) O[im][jd] = zero;
#pragma unroll
    for (int r = 0; r < 8; ++r) { rowm[im][r] = -3.0e38f; rowl[im][r] = 0.f; }
  }

  for (int kv = 0; kv < TT; kv += 64) {
    // ---- S = Q @ K^T : B-matrix lane = key column, d values in table order
    v8f S[4][4];
#pragma unroll
    for (int jn = 0; jn < 4; ++jn) {
      BF16x16 bK[2];
#pragma unroll
      for (int ks = 0; ks < 2; ++ks) {
        const __bf16* p = Kbase + (size_t)(kv + jn * 16 + l16) * HD + ks * 32 + half * 8;
        bK[ks].h[0] = *(const v8bf*)(p);
        bK[ks].h[1] = *(const v8bf*)(p + 16);
      }
#pragma unroll
      for (int im = 0; im < 4; ++im) {
        v8f s = wmma_bf16(aQ[im][0], bK[0], zero);
        S[im][jn] = wmma_bf16(aQ[im][1], bK[1], s);
      }
    }

    // ---- online softmax; stage P (bf16) into LDS in row-major A layout
#pragma unroll
    for (int im = 0; im < 4; ++im) {
#pragma unroll
      for (int r = 0; r < 8; ++r) {
        float s0 = S[im][0][r] * scale, s1 = S[im][1][r] * scale;
        float s2 = S[im][2][r] * scale, s3 = S[im][3][r] * scale;
        float mx = fmaxf(fmaxf(s0, s1), fmaxf(s2, s3));
#pragma unroll
        for (int off = 8; off >= 1; off >>= 1)
          mx = fmaxf(mx, __shfl_xor(mx, off, 32));   // rows live in 16-lane halves
        const float mnew = fmaxf(rowm[im][r], mx);
        const float alpha = __expf(rowm[im][r] - mnew);
        rowm[im][r] = mnew;
        float p0 = __expf(s0 - mnew), p1 = __expf(s1 - mnew);
        float p2 = __expf(s2 - mnew), p3 = __expf(s3 - mnew);
        float ps = p0 + p1 + p2 + p3;
#pragma unroll
        for (int off = 8; off >= 1; off >>= 1) ps += __shfl_xor(ps, off, 32);
        rowl[im][r] = rowl[im][r] * alpha + ps;
#pragma unroll
        for (int jd = 0; jd < 4; ++jd) O[im][jd][r] *= alpha;
        const int lrow = im * 16 + half * 8 + r;
        lds[lrow * 72 +  0 + l16] = (__bf16)p0;
        lds[lrow * 72 + 16 + l16] = (__bf16)p1;
        lds[lrow * 72 + 32 + l16] = (__bf16)p2;
        lds[lrow * 72 + 48 + l16] = (__bf16)p3;
      }
    }

    // ---- O += P @ V : V transposed so B-fragments are contiguous along keys
#pragma unroll
    for (int jd = 0; jd < 4; ++jd) {
      BF16x16 bV[2];
#pragma unroll
      for (int ks = 0; ks < 2; ++ks) {
        const __bf16* p = Vbase + (size_t)(jd * 16 + l16) * TT + kv + ks * 32 + half * 8;
        bV[ks].h[0] = *(const v8bf*)(p);
        bV[ks].h[1] = *(const v8bf*)(p + 16);
      }
#pragma unroll
      for (int im = 0; im < 4; ++im) {
        BF16x16 aP[2];
#pragma unroll
        for (int ks = 0; ks < 2; ++ks) {
          const __bf16* p = lds + (im * 16 + l16) * 72 + ks * 32 + half * 8;
          aP[ks].h[0] = *(const v8bf*)(p);
          aP[ks].h[1] = *(const v8bf*)(p + 16);
        }
        v8f o = wmma_bf16(aP[0], bV[0], O[im][jd]);
        O[im][jd] = wmma_bf16(aP[1], bV[1], o);
      }
    }
  }

  // ---- epilogue: normalize, store to [B,T,C] (C = H*64)
#pragma unroll
  for (int im = 0; im < 4; ++im)
#pragma unroll
    for (int jd = 0; jd < 4; ++jd)
#pragma unroll
      for (int r = 0; r < 8; ++r) {
        const int q = q0 + im * 16 + half * 8 + r;
        const float v = O[im][jd][r] / rowl[im][r];
        Aout[(size_t)(b * TT + q) * CC + h * HD + jd * 16 + l16] = (__bf16)v;
      }
}

// ---------------------------------------------------------------------------
extern "C" void kernel_launch(void* const* d_in, const int* in_sizes, int n_in,
                              void* d_out, int out_size, void* d_ws, size_t ws_size,
                              hipStream_t stream) {
  const float* hs = (const float*)d_in[0];
  const float* Wq = (const float*)d_in[1];
  const float* Wk = (const float*)d_in[2];
  const float* Wv = (const float*)d_in[3];
  const float* Wo = (const float*)d_in[4];
  const float* bo = (const float*)d_in[5];

  __bf16* ws = (__bf16*)d_ws;
  size_t off = 0;
  __bf16* Xb   = ws + off; off += (size_t)MROWS * CC;   // hidden bf16
  __bf16* Wqb  = ws + off; off += (size_t)CC * CC;
  __bf16* Wkb  = ws + off; off += (size_t)CC * CC;
  __bf16* Wvb  = ws + off; off += (size_t)CC * CC;
  __bf16* Wob  = ws + off; off += (size_t)CC * CC;
  __bf16* Qhd  = ws + off; off += (size_t)MROWS * CC;   // [B,H,T,D]
  __bf16* Khd  = ws + off; off += (size_t)MROWS * CC;   // [B,H,T,D]
  __bf16* Vtd  = ws + off; off += (size_t)MROWS * CC;   // [B,H,D,T]
  __bf16* Aout = ws + off; off += (size_t)MROWS * CC;   // [B,T,C]

  cast_f32_bf16<<<1024, 256, 0, stream>>>(hs, Xb, MROWS * CC);
  cast_f32_bf16<<<256, 256, 0, stream>>>(Wq, Wqb, CC * CC);
  cast_f32_bf16<<<256, 256, 0, stream>>>(Wk, Wkb, CC * CC);
  cast_f32_bf16<<<256, 256, 0, stream>>>(Wv, Wvb, CC * CC);
  cast_f32_bf16<<<256, 256, 0, stream>>>(Wo, Wob, CC * CC);

  // (MROWS/64) * (CC/64) = 2560 waves, 8 waves/block -> 320 blocks
  gemm_bf16_64x64<<<320, 256, 0, stream>>>(Xb, Wqb, Qhd, nullptr, 0);
  gemm_bf16_64x64<<<320, 256, 0, stream>>>(Xb, Wkb, Khd, nullptr, 0);
  gemm_bf16_64x64<<<320, 256, 0, stream>>>(Xb, Wvb, Vtd, nullptr, 1);

  // B*H*(T/64) = 2560 waves, 4 waves/block -> 640 blocks
  flash_attn<<<640, 128, 0, stream>>>(Qhd, Khd, Vtd, Aout);

  gemm_bf16_64x64<<<320, 256, 0, stream>>>(Aout, Wob, d_out, bo, 2);
}
